// DeformTransLayer_57741540327459
// MI455X (gfx1250) — matrix operand order repaired and checked by reading, not compiled, not run on hardware
//
#include <hip/hip_runtime.h>
#include <stdint.h>

// ---------------- problem constants (from reference) ----------------
#define N_    2
#define LQ_   5440
#define C_    256
#define H_    4
#define L_    4
#define P_    8
#define LIN_  5440            // 64*64+32*32+16*16+8*8
#define NTOK  (N_ * LQ_)      // 10880 (== N_*LIN_ too)
#define KD    256             // K of every GEMM

// ---------------- WMMA types (CDNA5, wave32) ----------------
typedef __attribute__((ext_vector_type(16))) __bf16 v16bf;
typedef __attribute__((ext_vector_type(8)))  float  v8f;

union FragBF { uint32_t u[8]; uint4 q[2]; v16bf v; };
union AccF   { float    f[8]; v8f   v; };

// round-to-nearest-even fp32 -> bf16, packed pair (lo = even elem, hi = odd)
__device__ __forceinline__ uint32_t pack_bf16(float lo, float hi) {
  uint32_t a = __float_as_uint(lo);
  uint32_t b = __float_as_uint(hi);
  a = (a + (0x7FFFu + ((a >> 16) & 1u))) >> 16;
  b = (b + (0x7FFFu + ((b >> 16) & 1u))) >> 16;
  return (b << 16) | (a & 0xFFFFu);
}

// ---------------- prep: fused add + bf16 pair pack ----------------
__global__ void __launch_bounds__(256) pack_add_k(const float2* __restrict__ a,
                                                  const float2* __restrict__ b,
                                                  uint32_t* __restrict__ o, int n) {
  int i = blockIdx.x * blockDim.x + threadIdx.x;
  if (i < n) { float2 x = a[i], y = b[i]; o[i] = pack_bf16(x.x + y.x, x.y + y.y); }
}

__global__ void __launch_bounds__(256) pack_k(const float2* __restrict__ a,
                                              uint32_t* __restrict__ o, int n) {
  int i = blockIdx.x * blockDim.x + threadIdx.x;
  if (i < n) { float2 x = a[i]; o[i] = pack_bf16(x.x, x.y); }
}

// weights (K=256 x Nw, row-major f32) -> WMMA B-fragment order:
//   o[((tileN*8 + kb)*32 + lane)*8 + v] = pack(W[k0][n], W[k0+1][n])
//   with n = tileN*16 + (lane&15), g = lane>>4, k0 = kb*32 + g*16 + 2v
// so the GEMM loads each B fragment as two contiguous b128s.
__global__ void __launch_bounds__(256) pack_w_frag_k(const float* __restrict__ w,
                                                     uint32_t* __restrict__ o,
                                                     int Nw, int total) {
  int i = blockIdx.x * blockDim.x + threadIdx.x;
  if (i >= total) return;
  const int v     = i & 7;
  const int lane  = (i >> 3) & 31;
  const int kb    = (i >> 8) & 7;
  const int tileN = i >> 11;
  const int n  = tileN * 16 + (lane & 15);
  const int g  = lane >> 4;
  const int k0 = kb * 32 + g * 16 + 2 * v;
  o[i] = pack_bf16(w[(size_t)k0 * Nw + n], w[(size_t)(k0 + 1) * Nw + n]);
}

// ---------------- bf16 WMMA GEMM: C(M x Nw) = A(M x 256) * B(256 x Nw) + bias ----
// A: bf16 pair-packed rows (128 u32 per row). B: fragment-ordered (see pack_w_frag_k).
// Each wave computes a 64x16 tile (4 M-subtiles sharing one B fragment per K-step).
// Per K-step: issue all 10 b128 loads (B + 4 A-subtiles into distinct registers),
// then 4 WMMAs, so the loads drain with partial s_wait_loadcnt instead of full
// barriers. Block = 2 waves = 128 rows; M = 10880 -> gridDim.x = 85 exactly.
__global__ void __launch_bounds__(64) gemm_bf16_k(const uint32_t* __restrict__ Apk,
                                                  const uint32_t* __restrict__ Bfr,
                                                  const float* __restrict__ bias,
                                                  float* __restrict__ Cf, int Nw) {
  const int lane  = threadIdx.x & 31;
  const int wave  = threadIdx.x >> 5;
  const int m0    = (blockIdx.x * 2 + wave) * 64;   // 64 rows per wave
  const int tileN = blockIdx.y;
  const int g     = lane >> 4;    // half-wave
  const int m     = lane & 15;
  const int ncol  = tileN * 16 + m;

  // A row pointer for this lane (subtile s adds an immediate 16*128 u32 offset)
  const uint32_t* arow = Apk + (size_t)(m0 + m) * (KD / 2) + g * 8;
  // B fragment pointer for this lane (K-step kb adds an immediate 256 u32 offset)
  const uint32_t* brow = Bfr + (size_t)tileN * 2048 + lane * 8;

  AccF acc[4];
#pragma unroll
  for (int s = 0; s < 4; ++s)
#pragma unroll
    for (int r = 0; r < 8; ++r) acc[s].f[r] = 0.0f;

#pragma unroll
  for (int kb = 0; kb < 8; ++kb) {            // K = kb*32 .. kb*32+31
    FragBF b, a[4];
    b.q[0] = *(const uint4*)(brow + kb * 256);
    b.q[1] = *(const uint4*)(brow + kb * 256 + 4);
#pragma unroll
    for (int s = 0; s < 4; ++s) {
      const uint32_t* ar = arow + (size_t)s * 16 * (KD / 2);
      a[s].q[0] = *(const uint4*)(ar + kb * 16);       // K: kb*32 + g*16 + [0..7]
      a[s].q[1] = *(const uint4*)(ar + kb * 16 + 4);   // K: kb*32 + g*16 + 8 + [0..7]
    }
#pragma unroll
    for (int s = 0; s < 4; ++s) {
      acc[s].v = __builtin_amdgcn_wmma_f32_16x16x32_bf16(
          /*neg_a=*/false, a[s].v, /*neg_b=*/false, b.v,
          /*c_mod=*/(short)0, acc[s].v, /*reuse_a=*/false, /*reuse_b=*/false);
    }
  }

  const float bs = bias[ncol];
  // C/D layout: VGPR r -> M = r + 8*g, N = lane&15
#pragma unroll
  for (int s = 0; s < 4; ++s) {
#pragma unroll
    for (int r = 0; r < 8; ++r) {
      const int row = m0 + s * 16 + r + 8 * g;
      Cf[(size_t)row * Nw + ncol] = acc[s].f[r] + bs;
    }
  }
}

// ---------------- deformable sampling: softmax + bilinear gather ----------------
// one wave per (token, head); lane p owns point p (L=4 levels * P=8 points)
__global__ void __launch_bounds__(256) sample_k(const float* __restrict__ valueF,  // (N*LIN, 256)
                                                const float* __restrict__ offsF,   // (NTOK, 256)
                                                const float* __restrict__ attnL,   // (NTOK, 128)
                                                const float* __restrict__ refp,    // (NTOK, 4, 2)
                                                uint32_t* __restrict__ outPk) {    // (NTOK, 128) bf16 pairs
  const int lane = threadIdx.x & 31;
  const int wid  = blockIdx.x * 8 + (threadIdx.x >> 5);
  const int h    = wid & 3;
  const int tok  = wid >> 2;
  const int n    = tok / LQ_;

  // --- softmax over the 32 (level,point) logits of this (token, head) ---
  float logit = attnL[(size_t)tok * 128 + h * 32 + lane];
  float mx = logit;
#pragma unroll
  for (int o = 16; o; o >>= 1) mx = fmaxf(mx, __shfl_xor(mx, o, 32));
  float e = __expf(logit - mx);
  float s = e;
#pragma unroll
  for (int o = 16; o; o >>= 1) s += __shfl_xor(s, o, 32);
  const float aw = e / s;

  // --- per-lane point geometry ---
  const int   l   = lane >> 3;          // level
  const int   Wl  = 64 >> l;            // square levels: H == W
  const float Wf  = (float)Wl;
  const int start = (l == 0) ? 0 : ((l == 1) ? 4096 : ((l == 2) ? 5120 : 5376));

  const float ox = offsF[(size_t)tok * 256 + h * 64 + 2 * lane];
  const float oy = offsF[(size_t)tok * 256 + h * 64 + 2 * lane + 1];
  const float rx = refp[((size_t)tok * L_ + l) * 2 + 0];
  const float ry = refp[((size_t)tok * L_ + l) * 2 + 1];
  // grid = 2*loc-1; gx = (grid_x+1)*W/2 - 0.5 = loc_x*W - 0.5
  const float gx = (rx + ox / Wf) * Wf - 0.5f;
  const float gy = (ry + oy / Wf) * Wf - 0.5f;
  const float x0f = floorf(gx), y0f = floorf(gy);
  const float fx = gx - x0f, fy = gy - y0f;
  const int x0 = (int)x0f, y0 = (int)y0f;

  int   ix[4];
  float cw[4];
#pragma unroll
  for (int c = 0; c < 4; ++c) {
    const int yy = y0 + (c >> 1), xx = x0 + (c & 1);
    const float wv = ((c >> 1) ? fy : 1.0f - fy) * ((c & 1) ? fx : 1.0f - fx);
    const bool valid = (yy >= 0) & (yy < Wl) & (xx >= 0) & (xx < Wl);
    const int yc = min(max(yy, 0), Wl - 1), xc = min(max(xx, 0), Wl - 1);
    ix[c] = ((n * LIN_ + start + yc * Wl + xc) << 8) + h * 64;  // float index of row chunk
    cw[c] = valid ? wv * aw : 0.0f;
  }

  // --- cooperative gather: 32 lanes sweep the 64-dim head vector (2 f32/lane) ---
  float ax = 0.0f, ay = 0.0f;
  const int d = 2 * lane;
  for (int p = 0; p < 32; ++p) {
#pragma unroll
    for (int c = 0; c < 4; ++c) {
      const int   gi = __shfl(ix[c], p, 32);
      const float w  = __shfl(cw[c], p, 32);
      const float2 v = *(const float2*)(valueF + gi + d);
      ax += w * v.x;
      ay += w * v.y;
    }
  }
  outPk[(size_t)tok * 128 + h * 32 + lane] = pack_bf16(ax, ay);
}

// ---------------- residual + layernorm: one wave per row ----------------
__global__ void __launch_bounds__(256) ln_k(const float* __restrict__ src,
                                            const float* __restrict__ attnOut,
                                            const float* __restrict__ g,
                                            const float* __restrict__ b,
                                            float* __restrict__ out) {
  const int lane = threadIdx.x & 31;
  const int row  = blockIdx.x * 8 + (threadIdx.x >> 5);
  const float* sp = src + (size_t)row * 256;
  const float* ap = attnOut + (size_t)row * 256;
  float x[8];
  float s = 0.0f;
#pragma unroll
  for (int i = 0; i < 8; ++i) {
    const int j = i * 32 + lane;
    x[i] = sp[j] + ap[j];
    s += x[i];
  }
#pragma unroll
  for (int o = 16; o; o >>= 1) s += __shfl_xor(s, o, 32);
  const float mean = s * (1.0f / 256.0f);
  float vs = 0.0f;
#pragma unroll
  for (int i = 0; i < 8; ++i) { const float dd = x[i] - mean; vs += dd * dd; }
#pragma unroll
  for (int o = 16; o; o >>= 1) vs += __shfl_xor(vs, o, 32);
  const float inv = rsqrtf(vs * (1.0f / 256.0f) + 1e-5f);
  float* op = out + (size_t)row * 256;
#pragma unroll
  for (int i = 0; i < 8; ++i) {
    const int j = i * 32 + lane;
    op[j] = (x[i] - mean) * inv * g[j] + b[j];
  }
}

// ---------------- host launch ----------------
extern "C" void kernel_launch(void* const* d_in, const int* in_sizes, int n_in,
                              void* d_out, int out_size, void* d_ws, size_t ws_size,
                              hipStream_t stream) {
  const float* src  = (const float*)d_in[0];
  const float* refp = (const float*)d_in[1];
  const float* key  = (const float*)d_in[2];
  const float* qpos = (const float*)d_in[5];
  const float* Wv   = (const float*)d_in[6];
  const float* bv   = (const float*)d_in[7];
  const float* Wo   = (const float*)d_in[8];
  const float* bo   = (const float*)d_in[9];
  const float* Wa   = (const float*)d_in[10];
  const float* ba   = (const float*)d_in[11];
  const float* Wp   = (const float*)d_in[12];
  const float* bp   = (const float*)d_in[13];
  const float* lng  = (const float*)d_in[14];
  const float* lnb  = (const float*)d_in[15];
  float* out = (float*)d_out;

  // workspace partition (all 256B-aligned)
  uint8_t* ws = (uint8_t*)d_ws;
  auto alloc = [&](size_t bytes) -> void* {
    void* p = (void*)ws;
    ws += (bytes + 255) & ~(size_t)255;
    return p;
  };
  uint32_t* qpk     = (uint32_t*)alloc((size_t)NTOK * 128 * 4);   // bf16(q) pairs
  uint32_t* keypk   = (uint32_t*)alloc((size_t)NTOK * 128 * 4);   // bf16(key) pairs
  uint32_t* WvFr    = (uint32_t*)alloc((size_t)16 * 2048 * 4);    // fragment-ordered weights
  uint32_t* WoFr    = (uint32_t*)alloc((size_t)16 * 2048 * 4);
  uint32_t* WaFr    = (uint32_t*)alloc((size_t)8  * 2048 * 4);
  uint32_t* WpFr    = (uint32_t*)alloc((size_t)16 * 2048 * 4);
  float*    valueF  = (float*)alloc((size_t)NTOK * 256 * 4);
  float*    offsF   = (float*)alloc((size_t)NTOK * 256 * 4);
  float*    attnLg  = (float*)alloc((size_t)NTOK * 128 * 4);
  uint32_t* outPk   = (uint32_t*)alloc((size_t)NTOK * 128 * 4);
  float*    attnOut = (float*)alloc((size_t)NTOK * 256 * 4);

  const int npairsTok = NTOK * 128;

  // 1) prep / bf16 packing
  pack_add_k<<<(npairsTok + 255) / 256, 256, 0, stream>>>(
      (const float2*)src, (const float2*)qpos, qpk, npairsTok);
  pack_k<<<(npairsTok + 255) / 256, 256, 0, stream>>>(
      (const float2*)key, keypk, npairsTok);
  pack_w_frag_k<<<(16 * 2048 + 255) / 256, 256, 0, stream>>>(Wv, WvFr, 256, 16 * 2048);
  pack_w_frag_k<<<(16 * 2048 + 255) / 256, 256, 0, stream>>>(Wo, WoFr, 256, 16 * 2048);
  pack_w_frag_k<<<(8 * 2048 + 255) / 256, 256, 0, stream>>>(Wa, WaFr, 128, 8 * 2048);
  pack_w_frag_k<<<(16 * 2048 + 255) / 256, 256, 0, stream>>>(Wp, WpFr, 256, 16 * 2048);

  // 2) WMMA GEMMs: value, offsets, attention logits (128 rows per block)
  const int mBlk = NTOK / 128;  // 85
  gemm_bf16_k<<<dim3(mBlk, 16), 64, 0, stream>>>(keypk, WvFr, bv, valueF, 256);
  gemm_bf16_k<<<dim3(mBlk, 16), 64, 0, stream>>>(qpk, WoFr, bo, offsF, 256);
  gemm_bf16_k<<<dim3(mBlk, 8),  64, 0, stream>>>(qpk, WaFr, ba, attnLg, 128);

  // 3) softmax + bilinear sampling + head-weighted sum -> bf16 rows
  sample_k<<<(NTOK * H_) / 8, 256, 0, stream>>>(valueF, offsF, attnLg, refp, outPk);

  // 4) output projection (WMMA) then residual + layernorm
  gemm_bf16_k<<<dim3(mBlk, 16), 64, 0, stream>>>(outPk, WpFr, bp, attnOut, 256);
  ln_k<<<NTOK / 8, 256, 0, stream>>>(src, attnOut, lng, lnb, out);
}
